// NovaIgnitionLM_9526237462633
// MI455X (gfx1250) — matrix operand order repaired and checked by reading, not compiled, not run on hardware
//
#include <hip/hip_runtime.h>
#include <hip/hip_bf16.h>
#include <cstdint>
#include <cstddef>

// ---------------------------------------------------------------------------
// Types for CDNA5 WMMA (gfx1250, wave32)
// ---------------------------------------------------------------------------
typedef __bf16 bf16_t;
typedef bf16_t v16bf __attribute__((ext_vector_type(16)));
typedef bf16_t v8bf  __attribute__((ext_vector_type(8)));
typedef float  v8f   __attribute__((ext_vector_type(8)));

#define TOK    2048   // B*T
#define DMODEL 768
#define NHEAD  8
#define DHEAD  64
#define TOTD   512    // H*DH
#define WINSZ  32
#define NEXP   4
#define EDIM   2048
#define VOCAB  32000

// ---------------------------------------------------------------------------
// Embedding gather + scale (D^-0.5)
// ---------------------------------------------------------------------------
__global__ void embed_kernel(const int* __restrict__ x,
                             const float* __restrict__ emb,
                             float* __restrict__ h) {
    int t   = blockIdx.x;
    int tid = threadIdx.x;
    int tok = x[t];
    const float* ep = emb + (size_t)tok * DMODEL;
    float*       hp = h   + (size_t)t   * DMODEL;
    for (int d = tid; d < DMODEL; d += 256)
        hp[d] = ep[d] * 0.03608439182435161f;   // 1/sqrt(768)
}

// ---------------------------------------------------------------------------
// LayerNorm over D=768; optionally writes f32 and/or bf16 results.
// One 256-thread block per token.
// ---------------------------------------------------------------------------
__global__ void ln_kernel(const float* __restrict__ x,
                          const float* __restrict__ g,
                          const float* __restrict__ b,
                          float* __restrict__ y32,
                          bf16_t* __restrict__ y16) {
    int t = blockIdx.x, tid = threadIdx.x;
    __shared__ float s1[256], s2[256];
    const float* xp = x + (size_t)t * DMODEL;
    float v0 = xp[tid], v1 = xp[tid + 256], v2 = xp[tid + 512];
    s1[tid] = v0 + v1 + v2;
    s2[tid] = v0 * v0 + v1 * v1 + v2 * v2;
    __syncthreads();
    for (int o = 128; o > 0; o >>= 1) {
        if (tid < o) { s1[tid] += s1[tid + o]; s2[tid] += s2[tid + o]; }
        __syncthreads();
    }
    float mean = s1[0] * (1.0f / DMODEL);
    float var  = s2[0] * (1.0f / DMODEL) - mean * mean;
    float rs   = rsqrtf(var + 1e-5f);
    float o0 = (v0 - mean) * rs * g[tid]       + b[tid];
    float o1 = (v1 - mean) * rs * g[tid + 256] + b[tid + 256];
    float o2 = (v2 - mean) * rs * g[tid + 512] + b[tid + 512];
    if (y32) {
        float* yp = y32 + (size_t)t * DMODEL;
        yp[tid] = o0; yp[tid + 256] = o1; yp[tid + 512] = o2;
    }
    if (y16) {
        bf16_t* yp = y16 + (size_t)t * DMODEL;
        yp[tid] = (bf16_t)o0; yp[tid + 256] = (bf16_t)o1; yp[tid + 512] = (bf16_t)o2;
    }
}

// ---------------------------------------------------------------------------
// BitNet weight quantization: per row s = max(mean|w|,1e-5),
// w_eff = clip(round(w/s),-1,1)*s, stored bf16.
// One 256-thread block per output row (N rows, K cols).
// ---------------------------------------------------------------------------
__global__ void quant_kernel(const float* __restrict__ w,
                             bf16_t* __restrict__ wq, int K) {
    int n = blockIdx.x, tid = threadIdx.x;
    const float* wp = w + (size_t)n * K;
    __shared__ float red[256];
    float s = 0.f;
    for (int k = tid; k < K; k += 256) s += fabsf(wp[k]);
    red[tid] = s; __syncthreads();
    for (int o = 128; o > 0; o >>= 1) {
        if (tid < o) red[tid] += red[tid + o];
        __syncthreads();
    }
    float scale = fmaxf(red[0] / (float)K, 1e-5f);
    float inv = 1.0f / scale;
    bf16_t* op = wq + (size_t)n * K;
    for (int k = tid; k < K; k += 256) {
        float q = rintf(wp[k] * inv);            // round-half-even like jnp.round
        q = fminf(fmaxf(q, -1.f), 1.f);
        op[k] = (bf16_t)(q * scale);
    }
}

// ---------------------------------------------------------------------------
// WMMA GEMM:  C[M,N] (f32) = A[M,K] (bf16 row-major) @ W[N,K]^T (bf16 row-major)
//
// Block = 128 threads = 4 waves splitting N (redundant loads fall on the
// small, cache-resident A operand; the big W stream is loaded exactly once
// per block). Each wave computes a 32x64 C tile: 2 M-subtiles x 4 N-subtiles
// = 8 v8f accumulators. Per K-step of 32: 2 A fragments reused 4x each, 4 B
// fragments reused 2x each -> 12 b128 loads per 8 v_wmma.
//
// A fragment (contiguous k, same order as B): lane l16=lane&15 holds row M=l16;
// half-group g=lane>>4 holds 16 contiguous K at k0+g*16 (two 16B chunks).
// B fragment (column n = weight row n): 16 contiguous bf16 at k0+g*16.
//
// NT=true marks C stores non-temporal (head GEMM: 262MB of logits written
// once and never re-read on-device; keep them from rinsing the 192MB L2).
// ---------------------------------------------------------------------------
template <bool NT>
__global__ void __launch_bounds__(128)
gemm_bf16_kernel(const bf16_t* __restrict__ A,
                 const bf16_t* __restrict__ W,
                 float* __restrict__ C,
                 int M, int N, int K) {
    const int wave = threadIdx.x >> 5;
    const int lane = threadIdx.x & 31;
    const int g    = lane >> 4;
    const int l16  = lane & 15;
    const int row0 = blockIdx.y * 32;                 // block tile: 32 rows
    const int col0 = blockIdx.x * 256 + wave * 64;    // wave tile: 64 cols

    const bf16_t* a_row0 = A + (size_t)(row0 + l16) * K;
    const bf16_t* a_row1 = a_row0 + (size_t)16 * K;

    v8f acc[2][4];
#pragma unroll
    for (int i = 0; i < 2; ++i)
#pragma unroll
        for (int j = 0; j < 4; ++j)
#pragma unroll
            for (int r = 0; r < 8; ++r) acc[i][j][r] = 0.0f;

    for (int k0 = 0; k0 < K; k0 += 32) {
        // ---- two A fragments (16x32 bf16 each): two 16B chunks per lane ----
        v16bf af0, af1;
        {
            v8bf lo0 = *(const v8bf*)(a_row0 + k0 + g * 16);
            v8bf hi0 = *(const v8bf*)(a_row0 + k0 + g * 16 + 8);
            v8bf lo1 = *(const v8bf*)(a_row1 + k0 + g * 16);
            v8bf hi1 = *(const v8bf*)(a_row1 + k0 + g * 16 + 8);
#pragma unroll
            for (int i = 0; i < 8; ++i) {
                af0[i] = lo0[i]; af0[i + 8] = hi0[i];
                af1[i] = lo1[i]; af1[i + 8] = hi1[i];
            }
        }

        // ---- 4 B fragments, each feeds 2 WMMAs ----
#pragma unroll
        for (int j = 0; j < 4; ++j) {
            const bf16_t* wrow =
                W + (size_t)(col0 + j * 16 + l16) * K + k0 + g * 16;
            // WGP-scope prefetch of the next K-tile of this W row
            // (locality=3 -> nearest cache level, global_prefetch_b8)
            __builtin_prefetch((const void*)(wrow + 32), 0, 3);
            v16bf bfr = *(const v16bf*)(wrow);   // 32B contiguous
            acc[0][j] = __builtin_amdgcn_wmma_f32_16x16x32_bf16(
                false, af0, false, bfr, (short)0, acc[0][j], false, false);
            acc[1][j] = __builtin_amdgcn_wmma_f32_16x16x32_bf16(
                false, af1, false, bfr, (short)0, acc[1][j], false, false);
        }
    }

    // ---- store: row = row0 + i*16 + g*8 + r, col = col0 + j*16 + l16 ----
#pragma unroll
    for (int i = 0; i < 2; ++i)
#pragma unroll
        for (int j = 0; j < 4; ++j)
#pragma unroll
            for (int r = 0; r < 8; ++r) {
                size_t row = (size_t)(row0 + i * 16 + g * 8 + r);
                float* cp =
                    C + row * (size_t)N + (size_t)(col0 + j * 16 + l16);
                if (NT) __builtin_nontemporal_store(acc[i][j][r], cp);
                else    *cp = acc[i][j][r];
            }
}

// ---------------------------------------------------------------------------
// Windowed causal attention, one wave per (token, head).
// qkv layout per token: [0,512)=q, [512,1024)=k, [1024,1536)=v.
// Writes bf16 output (input to the out-projection GEMM).
// ---------------------------------------------------------------------------
__global__ void attn_kernel(const float* __restrict__ qkv,
                            bf16_t* __restrict__ outb) {
    int t    = blockIdx.x;          // 0..2047 (b*1024 + tl)
    int h    = blockIdx.y;          // 0..7
    int lane = threadIdx.x;         // 0..31
    int b  = t >> 10;
    int tl = t & 1023;

    const float* qp = qkv + (size_t)t * 1536 + h * DHEAD;
    int  pos   = tl - (WINSZ - 1) + lane;
    bool valid = pos >= 0;
    int  pc    = valid ? pos : 0;
    const float* kp = qkv + (size_t)(b * 1024 + pc) * 1536 + 512 + h * DHEAD;

    float sc = 0.f;
#pragma unroll
    for (int d = 0; d < DHEAD; d += 4) {
        float4 qv = *(const float4*)(qp + d);
        float4 kv = *(const float4*)(kp + d);
        sc += qv.x * kv.x + qv.y * kv.y + qv.z * kv.z + qv.w * kv.w;
    }
    sc *= 0.125f;                     // DH^-0.5
    if (!valid) sc = -1e30f;

    float mx = sc;
    for (int o = 16; o > 0; o >>= 1) mx = fmaxf(mx, __shfl_xor(mx, o, 32));
    float e = valid ? __expf(sc - mx) : 0.f;
    float se = e;
    for (int o = 16; o > 0; o >>= 1) se += __shfl_xor(se, o, 32);
    float wt = e / se;

    float a0 = 0.f, a1 = 0.f;
    for (int w = 0; w < WINSZ; ++w) {
        float ww = __shfl(wt, w, 32);
        int p2  = tl - (WINSZ - 1) + w;
        int pcc = p2 < 0 ? 0 : p2;
        const float* vp = qkv + (size_t)(b * 1024 + pcc) * 1536 + 1024 + h * DHEAD;
        a0 += ww * vp[lane];
        a1 += ww * vp[lane + 32];
    }
    bf16_t* op = outb + (size_t)t * TOTD + h * DHEAD;
    op[lane]      = (bf16_t)a0;
    op[lane + 32] = (bf16_t)a1;
}

// ---------------------------------------------------------------------------
// Router: probs[t,e] = softmax_e( h[t] . router_w[e] ). One wave per token.
// ---------------------------------------------------------------------------
__global__ void router_kernel(const float* __restrict__ h32,
                              const float* __restrict__ rw,
                              float* __restrict__ probs) {
    int t = blockIdx.x, lane = threadIdx.x;
    const float* hp = h32 + (size_t)t * DMODEL;
    float acc0 = 0.f, acc1 = 0.f, acc2 = 0.f, acc3 = 0.f;
    for (int d = lane; d < DMODEL; d += 32) {
        float hv = hp[d];
        acc0 += hv * rw[d];
        acc1 += hv * rw[DMODEL + d];
        acc2 += hv * rw[2 * DMODEL + d];
        acc3 += hv * rw[3 * DMODEL + d];
    }
    for (int o = 16; o > 0; o >>= 1) {
        acc0 += __shfl_xor(acc0, o, 32);
        acc1 += __shfl_xor(acc1, o, 32);
        acc2 += __shfl_xor(acc2, o, 32);
        acc3 += __shfl_xor(acc3, o, 32);
    }
    if (lane == 0) {
        float mx = fmaxf(fmaxf(acc0, acc1), fmaxf(acc2, acc3));
        float e0 = __expf(acc0 - mx), e1 = __expf(acc1 - mx);
        float e2 = __expf(acc2 - mx), e3 = __expf(acc3 - mx);
        float s  = e0 + e1 + e2 + e3;
        float* pp = probs + (size_t)t * NEXP;
        pp[0] = e0 / s; pp[1] = e1 / s; pp[2] = e2 / s; pp[3] = e3 / s;
    }
}

// ---------------------------------------------------------------------------
// Elementwise kernels
// ---------------------------------------------------------------------------
__global__ void silu_bf16_kernel(const float* __restrict__ x,
                                 bf16_t* __restrict__ y, size_t n) {
    size_t i = (size_t)blockIdx.x * 256 + threadIdx.x;
    if (i < n) {
        float v = x[i];
        y[i] = (bf16_t)(v / (1.f + __expf(-v)));
    }
}

// weighted[t, e*ED+j] = silu(eh[t, e*ED+j]) * probs[t,e], bf16
__global__ void weighted_kernel(const float* __restrict__ eh,
                                const float* __restrict__ probs,
                                bf16_t* __restrict__ y, size_t n) {
    size_t i = (size_t)blockIdx.x * 256 + threadIdx.x;
    if (i < n) {
        size_t t = i >> 13;                 // / (NE*ED = 8192)
        int idx  = (int)(i & 8191);
        int e    = idx >> 11;               // / 2048
        float v  = eh[i];
        float sl = v / (1.f + __expf(-v));
        y[i] = (bf16_t)(sl * probs[t * NEXP + e]);
    }
}

__global__ void add2_kernel(float* __restrict__ x,
                            const float* __restrict__ a, size_t n) {
    size_t i = (size_t)blockIdx.x * 256 + threadIdx.x;
    if (i < n) x[i] += a[i];
}

__global__ void add3_kernel(float* __restrict__ x,
                            const float* __restrict__ a,
                            const float* __restrict__ b, size_t n) {
    size_t i = (size_t)blockIdx.x * 256 + threadIdx.x;
    if (i < n) x[i] += a[i] + b[i];
}

// ---------------------------------------------------------------------------
// Host-side orchestration
// ---------------------------------------------------------------------------
static inline void launch_gemm(const bf16_t* A, const bf16_t* W, float* C,
                               int M, int N, int K, bool nt, hipStream_t s) {
    dim3 grid(N / 256, M / 32);
    if (nt) gemm_bf16_kernel<true ><<<grid, 128, 0, s>>>(A, W, C, M, N, K);
    else    gemm_bf16_kernel<false><<<grid, 128, 0, s>>>(A, W, C, M, N, K);
}

extern "C" void kernel_launch(void* const* d_in, const int* in_sizes, int n_in,
                              void* d_out, int out_size, void* d_ws, size_t ws_size,
                              hipStream_t stream) {
    (void)in_sizes; (void)n_in; (void)out_size; (void)ws_size;

    // Input flattening assumption (setup_inputs insertion order):
    //   0: x, 1: embed, then 6 blocks x 15 tensors:
    //   [n1_g,n1_b,n2_g,n2_b, attn.{ln_g,ln_b,qkv_w,out_w},
    //    ffn.{ln_g,ln_b,sh1_w,sh2_w,up_w,down_w,router_w}]
    //   then lno_g, lno_b, head_w.
    const int* xin = (const int*)d_in[0];
    const float* embed = (const float*)d_in[1];
    const int NB = 6, PB = 15, BASE = 2;
    const float* lno_g  = (const float*)d_in[BASE + NB * PB + 0];
    const float* lno_b  = (const float*)d_in[BASE + NB * PB + 1];
    const float* head_w = (const float*)d_in[BASE + NB * PB + 2];

    // -------- workspace layout --------
    char* ws = (char*)d_ws;
    size_t off = 0;
    auto alloc = [&](size_t bytes) {
        size_t o = off;
        off += (bytes + 255) & ~(size_t)255;
        return o;
    };
    const size_t M = TOK;
    float*  h      = (float*) (ws + alloc(M * DMODEL * 4));       // residual stream
    float*  t1     = (float*) (ws + alloc(M * DMODEL * 4));       // LN #1 result
    float*  hb32   = (float*) (ws + alloc(M * DMODEL * 4));       // LN #2 f32 (router)
    float*  bufA   = (float*) (ws + alloc(M * 8192 * 4));         // qkv / m1 / eh
    bf16_t* bufB   = (bf16_t*)(ws + alloc(M * 8192 * 2));         // bf16 activations
    float*  c1     = (float*) (ws + alloc(M * DMODEL * 4));       // proj / shared
    float*  c2     = (float*) (ws + alloc(M * DMODEL * 4));       // dn
    bf16_t* wq     = (bf16_t*)(ws + alloc((size_t)VOCAB * DMODEL * 2)); // quantized W
    float*  probs  = (float*) (ws + alloc(M * NEXP * 4));

    bf16_t* a16  = bufB;                              // LN-out bf16 (3.1MB)
    bf16_t* at16 = bufB;                              // attention out bf16 (2MB)
    bf16_t* hb16 = bufB;                              // MoE LN-out bf16
    bf16_t* m1b  = bufB + (size_t)M * DMODEL;         // SiLU(shared mid) bf16
    bf16_t* wtd  = bufB;                              // weighted experts bf16

    const size_t nHD = M * DMODEL;

    // -------- embedding --------
    embed_kernel<<<dim3(TOK), 256, 0, stream>>>(xin, embed, h);

    // -------- transformer blocks --------
    for (int L = 0; L < NB; ++L) {
        const int p = BASE + L * PB;
        const float* n1_g  = (const float*)d_in[p + 0];
        const float* n1_b  = (const float*)d_in[p + 1];
        const float* n2_g  = (const float*)d_in[p + 2];
        const float* n2_b  = (const float*)d_in[p + 3];
        const float* aln_g = (const float*)d_in[p + 4];
        const float* aln_b = (const float*)d_in[p + 5];
        const float* qkv_w = (const float*)d_in[p + 6];
        const float* out_w = (const float*)d_in[p + 7];
        const float* fln_g = (const float*)d_in[p + 8];
        const float* fln_b = (const float*)d_in[p + 9];
        const float* sh1_w = (const float*)d_in[p + 10];
        const float* sh2_w = (const float*)d_in[p + 11];
        const float* up_w  = (const float*)d_in[p + 12];
        const float* dn_w  = (const float*)d_in[p + 13];
        const float* rt_w  = (const float*)d_in[p + 14];

        // ---- attention ----
        ln_kernel<<<dim3(TOK), 256, 0, stream>>>(h,  n1_g,  n1_b,  t1, nullptr);
        ln_kernel<<<dim3(TOK), 256, 0, stream>>>(t1, aln_g, aln_b, nullptr, a16);
        quant_kernel<<<dim3(3 * TOTD), 256, 0, stream>>>(qkv_w, wq, DMODEL);
        launch_gemm(a16, wq, bufA, M, 3 * TOTD, DMODEL, false, stream);   // qkv
        attn_kernel<<<dim3(TOK, NHEAD), 32, 0, stream>>>(bufA, at16);
        quant_kernel<<<dim3(DMODEL), 256, 0, stream>>>(out_w, wq, TOTD);
        launch_gemm(at16, wq, c1, M, DMODEL, TOTD, false, stream);        // out proj
        add2_kernel<<<dim3((int)((nHD + 255) / 256)), 256, 0, stream>>>(h, c1, nHD);

        // ---- MoE ----
        ln_kernel<<<dim3(TOK), 256, 0, stream>>>(h,  n2_g,  n2_b,  t1, nullptr);
        ln_kernel<<<dim3(TOK), 256, 0, stream>>>(t1, fln_g, fln_b, hb32, hb16);
        // shared expert
        quant_kernel<<<dim3(EDIM), 256, 0, stream>>>(sh1_w, wq, DMODEL);
        launch_gemm(hb16, wq, bufA, M, EDIM, DMODEL, false, stream);      // m1
        silu_bf16_kernel<<<dim3((int)((M * EDIM + 255) / 256)), 256, 0, stream>>>(
            bufA, m1b, M * EDIM);
        quant_kernel<<<dim3(DMODEL), 256, 0, stream>>>(sh2_w, wq, EDIM);
        launch_gemm(m1b, wq, c1, M, DMODEL, EDIM, false, stream);         // shared
        // router + experts
        router_kernel<<<dim3(TOK), 32, 0, stream>>>(hb32, rt_w, probs);
        quant_kernel<<<dim3(NEXP * EDIM), 256, 0, stream>>>(up_w, wq, DMODEL);
        launch_gemm(hb16, wq, bufA, M, NEXP * EDIM, DMODEL, false, stream); // eh
        weighted_kernel<<<dim3((int)((M * NEXP * EDIM + 255) / 256)), 256, 0, stream>>>(
            bufA, probs, wtd, M * NEXP * EDIM);
        quant_kernel<<<dim3(DMODEL), 256, 0, stream>>>(dn_w, wq, NEXP * EDIM);
        launch_gemm(wtd, wq, c2, M, DMODEL, NEXP * EDIM, false, stream);  // down
        add3_kernel<<<dim3((int)((nHD + 255) / 256)), 256, 0, stream>>>(h, c1, c2, nHD);
    }

    // -------- final LN + head (non-temporal logit stores) --------
    ln_kernel<<<dim3(TOK), 256, 0, stream>>>(h, lno_g, lno_b, nullptr, a16);
    quant_kernel<<<dim3(VOCAB), 256, 0, stream>>>(head_w, wq, DMODEL);
    launch_gemm(a16, wq, (float*)d_out, M, VOCAB, DMODEL, true, stream);
}
